// DKNet_42288247996638
// MI455X (gfx1250) — matrix operand order repaired and checked by reading, not compiled, not run on hardware
//
#include <hip/hip_runtime.h>
#include <hip/hip_bf16.h>
#include <math.h>

#define NPTS   100000
#define NB     4
#define KTOP   32
#define NI     128          // NB*KTOP instances
#define ODIM   16
#define WCOLS  337
#define MD     35           // merge tower width (2*OD+3)
#define MROWS  (NI*NI)      // 16384
#define OUTSTRIDE 100128    // NPTS + NI
#define EPSF   1e-5f

typedef __attribute__((ext_vector_type(16))) _Float16 v16h;
typedef __attribute__((ext_vector_type(8)))  float    v8f;

__device__ __forceinline__ float dk_sigmoid(float x) {
    // 1/(1+exp(-x)) with hardware v_exp + v_rcp (no precise-div expansion)
    return __builtin_amdgcn_rcpf(1.f + __expf(-x));
}

// ---------------------------------------------------------------------------
// Column stats: stats[c] = sum_n y[n,c]; stats[ncols+c] = sum_n y[n,c]^2
// One block per column; deterministic LDS tree reduce, no atomics.
__global__ void dk_colstats(const float* __restrict__ y, int nrows, int ncols,
                            float* __restrict__ stats) {
    __shared__ float s1[256], s2[256];
    const int c = blockIdx.x;
    const int tid = threadIdx.x;
    float a = 0.f, b = 0.f;
    for (int n = tid; n < nrows; n += 256) {
        float v = y[(size_t)n * ncols + c];
        a += v;
        b += v * v;
    }
    s1[tid] = a; s2[tid] = b;
    __syncthreads();
    for (int off = 128; off > 0; off >>= 1) {
        if (tid < off) { s1[tid] += s1[tid + off]; s2[tid] += s2[tid + off]; }
        __syncthreads();
    }
    if (tid == 0) { stats[c] = s1[0]; stats[ncols + c] = s2[0]; }
}

// ---------------------------------------------------------------------------
// Tower layer: y = (optional bn+relu of x) @ W (+ optional bias)
__global__ void dk_tower_layer(const float* __restrict__ xin,
                               const float* __restrict__ stats_in,
                               const float* __restrict__ W,     // 32 x Dout
                               const float* __restrict__ bias,  // Dout or null
                               float* __restrict__ yout,        // N x Dout
                               int Dout) {
    __shared__ float sW[32 * 32];
    __shared__ float sMean[32], sRstd[32];
    const int tid = threadIdx.x;
    for (int i = tid; i < 32 * Dout; i += blockDim.x) sW[i] = W[i];
    if (stats_in && tid < 32) {
        float m = stats_in[tid] * (1.0f / NPTS);
        float v = stats_in[32 + tid] * (1.0f / NPTS) - m * m;
        sMean[tid] = m;
        sRstd[tid] = rsqrtf(v + EPSF);
    }
    __syncthreads();

    const int n = blockIdx.x * blockDim.x + tid;
    if (n >= NPTS) return;
    float x[32];
    const float* xr = xin + (size_t)n * 32;
#pragma unroll
    for (int k = 0; k < 32; ++k) x[k] = xr[k];
    if (stats_in) {
#pragma unroll
        for (int k = 0; k < 32; ++k) {
            float t = (x[k] - sMean[k]) * sRstd[k];
            x[k] = t > 0.f ? t : 0.f;
        }
    }
    float* yr = yout + (size_t)n * Dout;
    for (int c = 0; c < Dout; ++c) {
        float acc = bias ? bias[c] : 0.f;
#pragma unroll
        for (int k = 0; k < 32; ++k) acc += x[k] * sW[k * Dout + c];
        yr[c] = acc;
    }
}

// ---------------------------------------------------------------------------
// Serial top-K NMS, fused argmax+suppression:
// After each winner, a single strided pass per thread both suppresses and
// recomputes the thread's running max; argmax is then a 1024-wide LDS reduce.
// Only live scores (s != -inf) need the distance test: dead points (other
// batches or already suppressed) stay -inf either way.
__global__ __launch_bounds__(1024)
void dk_topk_nms(const float* __restrict__ heat,
                 const float* __restrict__ coords,
                 const int* __restrict__ batch_idxs,
                 float* __restrict__ s,          // N scratch scores
                 int* __restrict__ topk_idx) {   // NI indices
    __shared__ float sv[1024];
    __shared__ int   si[1024];
    __shared__ float sc[3];
    const int tid = threadIdx.x;
    const float NEGINF = -__builtin_inff();
    for (int b = 0; b < NB; ++b) {
        // init pass: masked scores + per-thread max over its strided subset
        float best = NEGINF; int bi = 0x7fffffff;
        for (int n = tid; n < NPTS; n += 1024) {
            float v = (batch_idxs[n] == b) ? heat[n] : NEGINF;
            s[n] = v;
            if (v > best || (v == best && n < bi)) { best = v; bi = n; }
        }
        sv[tid] = best; si[tid] = bi;
        __syncthreads();
        for (int k = 0; k < KTOP; ++k) {
            // block argmax over the 1024 per-thread maxima
            for (int off = 512; off > 0; off >>= 1) {
                if (tid < off) {
                    float v2 = sv[tid + off]; int i2 = si[tid + off];
                    if (v2 > sv[tid] || (v2 == sv[tid] && i2 < si[tid])) {
                        sv[tid] = v2; si[tid] = i2;
                    }
                }
                __syncthreads();
            }
            if (tid == 0) {
                int idx = si[0];
                topk_idx[b * KTOP + k] = idx;
                sc[0] = coords[idx * 3 + 0];
                sc[1] = coords[idx * 3 + 1];
                sc[2] = coords[idx * 3 + 2];
            }
            __syncthreads();
            const float cx = sc[0], cy = sc[1], cz = sc[2];
            // fused suppression + local-max rebuild
            best = NEGINF; bi = 0x7fffffff;
            for (int n = tid; n < NPTS; n += 1024) {
                float v = s[n];
                if (v != NEGINF) {
                    float dx = coords[n * 3 + 0] - cx;
                    float dy = coords[n * 3 + 1] - cy;
                    float dz = coords[n * 3 + 2] - cz;
                    if (dx * dx + dy * dy + dz * dz < 0.09f) {
                        v = NEGINF;
                        s[n] = NEGINF;
                    }
                }
                if (v > best || (v == best && n < bi)) { best = v; bi = n; }
            }
            sv[tid] = best; si[tid] = bi;
            __syncthreads();
        }
    }
}

// ---------------------------------------------------------------------------
// Gather candidate arrays + build feat = [cand_kernel | cand_mask | ctr]
__global__ void dk_gather(const int* __restrict__ topk_idx,
                          const float* __restrict__ coords,
                          const float* __restrict__ maskF,
                          const float* __restrict__ kernF,
                          const int* __restrict__ batch_idxs,
                          float* __restrict__ ctr,         // NI x 3
                          float* __restrict__ ckern,       // NI x 16
                          int* __restrict__ cbatch,        // NI
                          float* __restrict__ feat) {      // NI x 35
    const int i = threadIdx.x;
    if (i >= NI) return;
    const int idx = topk_idx[i];
#pragma unroll
    for (int p = 0; p < 3; ++p) {
        float v = coords[idx * 3 + p];
        ctr[i * 3 + p] = v;
        feat[i * MD + 32 + p] = v;
    }
#pragma unroll
    for (int h = 0; h < ODIM; ++h) {
        float vk = kernF[(size_t)idx * ODIM + h];
        float vm = maskF[(size_t)idx * ODIM + h];
        ckern[i * ODIM + h] = vk;
        feat[i * MD + h] = vk;
        feat[i * MD + ODIM + h] = vm;
    }
    cbatch[i] = batch_idxs[idx];
}

// ---------------------------------------------------------------------------
// weights = cand_kernel @ Wwg + bwg   (NI x 337)
__global__ void dk_weight_gemm(const float* __restrict__ ckern,
                               const float* __restrict__ Wwg,   // 16 x 337
                               const float* __restrict__ bwg,
                               float* __restrict__ wts) {
    const int t = blockIdx.x * blockDim.x + threadIdx.x;
    if (t >= NI * WCOLS) return;
    const int i = t / WCOLS, j = t - i * WCOLS;
    float a = bwg[j];
#pragma unroll
    for (int h = 0; h < ODIM; ++h) a += ckern[i * ODIM + h] * Wwg[h * WCOLS + j];
    wts[t] = a;
}

// ---------------------------------------------------------------------------
// Build augmented f16 matrix: [mask_feats(16) | coords(3) | 1.0 | zeros(12)]
// The constant-1 column at K=19 lets the per-instance bias live in the A matrix.
__global__ void dk_build_aug(const float* __restrict__ maskF,
                             const float* __restrict__ coords,
                             _Float16* __restrict__ Aaug) {
    const int n = blockIdx.x * blockDim.x + threadIdx.x;
    if (n >= NPTS) return;
    _Float16* r = Aaug + (size_t)n * 32;
#pragma unroll
    for (int k = 0; k < ODIM; ++k) r[k] = (_Float16)maskF[(size_t)n * ODIM + k];
#pragma unroll
    for (int p = 0; p < 3; ++p) r[ODIM + p] = (_Float16)coords[n * 3 + p];
    r[19] = (_Float16)1.f;
#pragma unroll
    for (int k = 20; k < 32; ++k) r[k] = (_Float16)0.f;
}

// ---------------------------------------------------------------------------
// Dynamic mask head via WMMA f16->f32, transposed mapping:
//   A (16x32 f16) = fused instance weights^T, bias folded at K=19:
//       A[j][k<19] = W1[k][j];  A[j][19] = b1[j] - ctr . W1b[:,j];  else 0
//       lanes 0-15:  M=lane,    halves0-7 = K0..7,  halves8-15 = K16..23
//       lanes 16-31: M=lane-16, halves0-7 = K8..15, halves8-15 = K24..31
//   B (32x16 f16) = point tile (two 16B row chunks per lane: K 8h..8h+7, 16+8h..23+8h)
//   C = inline zero
//   D (16x16 f32): VGPR r: lanes0-15 -> M=j=r, N=pt=lane;
//                          lanes16-31 -> M=j=8+r, N=pt=lane-16
// Dot over hidden j = 8 in-register FMAs + one xor-16 shuffle.
__global__ __launch_bounds__(256)
void dk_mask_head(const _Float16* __restrict__ Aaug,
                  const float* __restrict__ wts,    // NI x 337
                  const float* __restrict__ ctr,    // NI x 3
                  float* __restrict__ dout) {
    const int lane = threadIdx.x & 31;
    const int wave = threadIdx.x >> 5;
    const int inst = blockIdx.y;
    const float* wrow = wts + (size_t)inst * WCOLS;
    const int n16 = lane & 15;
    const bool lowhalf = (lane < 16);

    // per-row fused bias (consumed only by low-half lanes at K=19)
    const int jrow = n16;                 // M = hidden index owned by this lane
    const float cx = ctr[inst * 3 + 0], cy = ctr[inst * 3 + 1], cz = ctr[inst * 3 + 2];
    float cbias = wrow[304 + jrow];
    cbias -= cx * wrow[(ODIM + 0) * 16 + jrow];
    cbias -= cy * wrow[(ODIM + 1) * 16 + jrow];
    cbias -= cz * wrow[(ODIM + 2) * 16 + jrow];

    // A matrix: instance weights transposed, bias at K=19
    v16h amat;
    const int kofflo = lowhalf ? 0 : 8;
#pragma unroll
    for (int jj = 0; jj < 8; ++jj) {
        const int k0 = kofflo + jj;       // <= 15, always a weight
        const int k1 = 16 + kofflo + jj;  // 16..23 (low half) or 24..31 (high half)
        amat[jj] = (_Float16)wrow[k0 * 16 + jrow];
        float a1;
        if (k1 < 19)       a1 = wrow[k1 * 16 + jrow];
        else if (k1 == 19) a1 = cbias;
        else               a1 = 0.f;
        amat[8 + jj] = (_Float16)a1;
    }
    // per-lane W2 for the 8 accumulator rows (j = r + 0 or r + 8)
    const int jbase = lowhalf ? 0 : 8;
    float w2v[8];
#pragma unroll
    for (int r = 0; r < 8; ++r) w2v[r] = wrow[320 + jbase + r];
    const float b2 = wrow[336];

    const int kbaseB = lowhalf ? 0 : 8;       // half-offset into Aaug row
    const int tile0 = (blockIdx.x * 8 + wave) * 128;

#pragma unroll
    for (int t = 0; t < 8; ++t) {
        const int m0 = tile0 + t * 16;
        const int m = m0 + n16;               // point handled by this lane
        v16h bmat;
        if (m < NPTS) {
            typedef __attribute__((ext_vector_type(8))) _Float16 v8h;
            const v8h* bp = (const v8h*)(Aaug + (size_t)m * 32 + kbaseB);
            bmat = __builtin_shufflevector(bp[0], bp[2], 0, 1, 2, 3, 4, 5, 6, 7, 8, 9, 10, 11, 12, 13, 14, 15);
        } else {
#pragma unroll
            for (int j = 0; j < 16; ++j) bmat[j] = (_Float16)0.f;
        }
        v8f acc = {};                          // C = 0 (bias lives in A)
        acc = __builtin_amdgcn_wmma_f32_16x16x32_f16(
                  false, amat, false, bmat, (short)0, acc, false, false);
        float partial = 0.f;
#pragma unroll
        for (int r = 0; r < 8; ++r) {
            float h = acc[r];
            h = h > 0.f ? h : 0.f;
            partial += h * w2v[r];
        }
        // combine hidden j=0..7 (lanes 0-15) with j=8..15 (lanes 16-31)
        float total = partial + __shfl_xor(partial, 16, 32);
        if (lowhalf && m < NPTS) {
            dout[(size_t)inst * OUTSTRIDE + m] = dk_sigmoid(total + b2);
        }
    }
}

// ---------------------------------------------------------------------------
// diff[a,b,c] = max(|feat[a,c] - feat[b,c]|, 1e-6)
__global__ void dk_diff(const float* __restrict__ feat, float* __restrict__ diff) {
    const int t = blockIdx.x * blockDim.x + threadIdx.x;
    if (t >= MROWS * MD) return;
    const int row = t / MD, cIdx = t - row * MD;
    const int a = row >> 7, b = row & 127;
    float d = fabsf(feat[a * MD + cIdx] - feat[b * MD + cIdx]);
    diff[t] = fmaxf(d, 1e-6f);
}

// ---------------------------------------------------------------------------
// Merge tower layer (35 -> 35): y = (optional bn+relu of x) @ W
__global__ void dk_merge_layer(const float* __restrict__ xin,
                               const float* __restrict__ stats_in,   // 70 or null
                               const float* __restrict__ W,          // 35 x 35
                               float* __restrict__ yout) {
    __shared__ float sW[MD * MD];
    __shared__ float sMean[MD], sRstd[MD];
    const int tid = threadIdx.x;
    for (int i = tid; i < MD * MD; i += 256) sW[i] = W[i];
    if (stats_in && tid < MD) {
        float m = stats_in[tid] * (1.f / MROWS);
        float v = stats_in[MD + tid] * (1.f / MROWS) - m * m;
        sMean[tid] = m;
        sRstd[tid] = rsqrtf(v + EPSF);
    }
    __syncthreads();
    const int row = blockIdx.x * 256 + tid;
    const float* xr = xin + (size_t)row * MD;
    float x[MD];
#pragma unroll
    for (int k = 0; k < MD; ++k) {
        float t = xr[k];
        if (stats_in) { t = (t - sMean[k]) * sRstd[k]; t = t > 0.f ? t : 0.f; }
        x[k] = t;
    }
    float* yr = yout + (size_t)row * MD;
    for (int cc = 0; cc < MD; ++cc) {
        float a = 0.f;
#pragma unroll
        for (int k = 0; k < MD; ++k) a += x[k] * sW[k * MD + cc];
        yr[cc] = a;
    }
}

// ---------------------------------------------------------------------------
// Merge output: sigmoid(relu(bn(x)) @ Wout + bout), batch-masked.
__global__ void dk_merge_out(const float* __restrict__ xin,
                             const float* __restrict__ stats_in,
                             const float* __restrict__ Wout,   // 35
                             const float* __restrict__ bout,   // 1
                             const int* __restrict__ cbatch,
                             float* __restrict__ dout) {
    __shared__ float sMean[MD], sRstd[MD], sWo[MD];
    const int tid = threadIdx.x;
    if (tid < MD) {
        float m = stats_in[tid] * (1.f / MROWS);
        float v = stats_in[MD + tid] * (1.f / MROWS) - m * m;
        sMean[tid] = m;
        sRstd[tid] = rsqrtf(v + EPSF);
        sWo[tid] = Wout[tid];
    }
    __syncthreads();
    const int row = blockIdx.x * 256 + tid;
    const float* xr = xin + (size_t)row * MD;
    float a = bout[0];
#pragma unroll
    for (int k = 0; k < MD; ++k) {
        float t = (xr[k] - sMean[k]) * sRstd[k];
        t = t > 0.f ? t : 0.f;
        a += t * sWo[k];
    }
    float sgm = dk_sigmoid(a);
    const int ia = row >> 7, ib = row & 127;
    if (cbatch[ia] != cbatch[ib]) sgm = 0.f;
    dout[(size_t)ia * OUTSTRIDE + NPTS + ib] = sgm;
}

// ---------------------------------------------------------------------------
extern "C" void kernel_launch(void* const* d_in, const int* in_sizes, int n_in,
                              void* d_out, int out_size, void* d_ws, size_t ws_size,
                              hipStream_t stream) {
    (void)in_sizes; (void)n_in; (void)out_size; (void)ws_size;
    const float* output_feats = (const float*)d_in[0];
    const float* coords       = (const float*)d_in[1];
    const float* heat         = (const float*)d_in[2];
    const int*   batch_idxs   = (const int*)d_in[3];
    const float* Wm           = (const float*)d_in[4];   // 3 x 32 x 32
    const float* Wm_out       = (const float*)d_in[5];   // 32 x 16
    const float* bm_out       = (const float*)d_in[6];
    const float* Wk           = (const float*)d_in[7];
    const float* Wk_out       = (const float*)d_in[8];
    const float* bk_out       = (const float*)d_in[9];
    const float* Wg           = (const float*)d_in[10];  // 3 x 35 x 35
    const float* Wg_out       = (const float*)d_in[11];  // 35 x 1
    const float* bg_out       = (const float*)d_in[12];
    const float* Wwg          = (const float*)d_in[13];  // 16 x 337
    const float* bwg          = (const float*)d_in[14];
    float* out = (float*)d_out;

    char* ws = (char*)d_ws;
    size_t off = 0;
    auto alloc = [&](size_t bytes) -> char* {
        char* p = ws + off;
        off += (bytes + 255) & ~(size_t)255;
        return p;
    };
    float*    yA    = (float*)alloc((size_t)NPTS * 32 * 4);
    float*    yB    = (float*)alloc((size_t)NPTS * 32 * 4);
    float*    maskF = (float*)alloc((size_t)NPTS * ODIM * 4);
    float*    kernF = (float*)alloc((size_t)NPTS * ODIM * 4);
    _Float16* Aaug  = (_Float16*)alloc((size_t)NPTS * 32 * 2);
    float*    sbuf  = (float*)alloc((size_t)NPTS * 4);
    int*      topk  = (int*)alloc(NI * 4);
    float*    cctr  = (float*)alloc(NI * 3 * 4);
    float*    ckern = (float*)alloc(NI * ODIM * 4);
    int*      cbat  = (int*)alloc(NI * 4);
    float*    wtsb  = (float*)alloc((size_t)NI * WCOLS * 4);
    float*    featb = (float*)alloc(NI * MD * 4);
    float*    mA    = (float*)alloc((size_t)MROWS * MD * 4);
    float*    mB    = (float*)alloc((size_t)MROWS * MD * 4);
    float*    stats = (float*)alloc(16 * 128 * 4);   // slots of 128 floats

    float* st0 = stats + 0 * 128;
    float* st1 = stats + 1 * 128;
    float* st2 = stats + 2 * 128;
    float* st3 = stats + 3 * 128;
    float* st4 = stats + 4 * 128;
    float* st5 = stats + 5 * 128;
    float* st6 = stats + 6 * 128;
    float* st7 = stats + 7 * 128;
    float* st8 = stats + 8 * 128;

    const int gN = (NPTS + 255) / 256;
    // mask tower
    dk_tower_layer<<<gN, 256, 0, stream>>>(output_feats, nullptr, Wm, nullptr, yA, 32);
    dk_colstats<<<32, 256, 0, stream>>>(yA, NPTS, 32, st0);
    dk_tower_layer<<<gN, 256, 0, stream>>>(yA, st0, Wm + 1024, nullptr, yB, 32);
    dk_colstats<<<32, 256, 0, stream>>>(yB, NPTS, 32, st1);
    dk_tower_layer<<<gN, 256, 0, stream>>>(yB, st1, Wm + 2048, nullptr, yA, 32);
    dk_colstats<<<32, 256, 0, stream>>>(yA, NPTS, 32, st2);
    dk_tower_layer<<<gN, 256, 0, stream>>>(yA, st2, Wm_out, bm_out, maskF, 16);
    // kernel tower
    dk_tower_layer<<<gN, 256, 0, stream>>>(output_feats, nullptr, Wk, nullptr, yA, 32);
    dk_colstats<<<32, 256, 0, stream>>>(yA, NPTS, 32, st3);
    dk_tower_layer<<<gN, 256, 0, stream>>>(yA, st3, Wk + 1024, nullptr, yB, 32);
    dk_colstats<<<32, 256, 0, stream>>>(yB, NPTS, 32, st4);
    dk_tower_layer<<<gN, 256, 0, stream>>>(yB, st4, Wk + 2048, nullptr, yA, 32);
    dk_colstats<<<32, 256, 0, stream>>>(yA, NPTS, 32, st5);
    dk_tower_layer<<<gN, 256, 0, stream>>>(yA, st5, Wk_out, bk_out, kernF, 16);

    // serial top-k NMS (fused argmax+suppression)
    dk_topk_nms<<<1, 1024, 0, stream>>>(heat, coords, batch_idxs, sbuf, topk);
    // gather candidates + feat
    dk_gather<<<1, NI, 0, stream>>>(topk, coords, maskF, kernF, batch_idxs,
                                    cctr, ckern, cbat, featb);
    // per-instance weight generator
    dk_weight_gemm<<<(NI * WCOLS + 255) / 256, 256, 0, stream>>>(ckern, Wwg, bwg, wtsb);

    // augmented f16 matrix for WMMA
    dk_build_aug<<<gN, 256, 0, stream>>>(maskF, coords, Aaug);
    // dynamic mask head (WMMA): grid (point tiles, instances)
    dim3 mgrid((NPTS + 1023) / 1024, NI);
    dk_mask_head<<<mgrid, 256, 0, stream>>>(Aaug, wtsb, cctr, out);

    // merge tower
    dk_diff<<<(MROWS * MD + 255) / 256, 256, 0, stream>>>(featb, mA);
    dk_merge_layer<<<MROWS / 256, 256, 0, stream>>>(mA, nullptr, Wg, mB);
    dk_colstats<<<MD, 256, 0, stream>>>(mB, MROWS, MD, st6);
    dk_merge_layer<<<MROWS / 256, 256, 0, stream>>>(mB, st6, Wg + 1225, mA);
    dk_colstats<<<MD, 256, 0, stream>>>(mA, MROWS, MD, st7);
    dk_merge_layer<<<MROWS / 256, 256, 0, stream>>>(mA, st7, Wg + 2450, mB);
    dk_colstats<<<MD, 256, 0, stream>>>(mB, MROWS, MD, st8);
    dk_merge_out<<<MROWS / 256, 256, 0, stream>>>(mB, st8, Wg_out, bg_out, cbat, out);
}
